// StructureGuidedContrastiveLoss_39608188404149
// MI455X (gfx1250) — matrix operand order repaired and checked by reading, not compiled, not run on hardware
//
#include <hip/hip_runtime.h>
#include <hip/hip_bf16.h>
#include <math.h>

#define NNODES 16384
#define DIM    128
#define TWO_N  32768
#define NEDGE  262144

#define QROWS  64      // query rows per block (4 waves x 16)
#define TKEYS  64      // keys staged per outer iteration
#define LDSB   136     // padded LDS row stride in bf16 elems (272B -> 4-bank skew)

typedef __attribute__((ext_vector_type(8)))  float  v8f;
typedef __attribute__((ext_vector_type(16))) __bf16 v16bf;
typedef __attribute__((ext_vector_type(8)))  __bf16 v8bf;
typedef __attribute__((ext_vector_type(4)))  __bf16 v4bf;

__device__ __forceinline__ float wave_reduce_sum(float v) {
  v += __shfl_xor(v, 1);
  v += __shfl_xor(v, 2);
  v += __shfl_xor(v, 4);
  v += __shfl_xor(v, 8);
  v += __shfl_xor(v, 16);
  return v;
}

__device__ __forceinline__ float half_reduce_sum(float v) {  // within 16 lanes
  v += __shfl_xor(v, 1);
  v += __shfl_xor(v, 2);
  v += __shfl_xor(v, 4);
  v += __shfl_xor(v, 8);
  return v;
}

__device__ __forceinline__ v16bf cat8(v8bf lo, v8bf hi) {
  return __builtin_shufflevector(lo, hi, 0, 1, 2, 3, 4, 5, 6, 7,
                                         8, 9, 10, 11, 12, 13, 14, 15);
}

// ------- 1) L2-normalize rows; emit fp32 plane + split-bf16 (hi/lo) planes --
__global__ void norm_kernel(const float* __restrict__ hv,
                            const float* __restrict__ hc,
                            float* __restrict__ hn,
                            __bf16* __restrict__ hhi,
                            __bf16* __restrict__ hlo) {
  int wave = threadIdx.x >> 5, lane = threadIdx.x & 31;
  int row = blockIdx.x * 8 + wave;
  const float* src = (row < NNODES) ? (hv + (size_t)row * DIM)
                                    : (hc + (size_t)(row - NNODES) * DIM);
  float4 x = ((const float4*)src)[lane];
  float ss = x.x * x.x + x.y * x.y + x.z * x.z + x.w * x.w;
  ss = wave_reduce_sum(ss);
  float inv = 1.0f / fmaxf(sqrtf(ss), 1e-12f);
  x.x *= inv; x.y *= inv; x.z *= inv; x.w *= inv;
  ((float4*)(hn + (size_t)row * DIM))[lane] = x;

  __bf16 h0 = (__bf16)x.x, h1 = (__bf16)x.y, h2 = (__bf16)x.z, h3 = (__bf16)x.w;
  __bf16 l0 = (__bf16)(x.x - (float)h0);
  __bf16 l1 = (__bf16)(x.y - (float)h1);
  __bf16 l2 = (__bf16)(x.z - (float)h2);
  __bf16 l3 = (__bf16)(x.w - (float)h3);
  v4bf vh = {h0, h1, h2, h3};
  v4bf vl = {l0, l1, l2, l3};
  *(v4bf*)(hhi + (size_t)row * DIM + 4 * lane) = vh;
  *(v4bf*)(hlo + (size_t)row * DIM + 4 * lane) = vl;
}

// ------- 2) self + positive-pair corrections, pos_sim (exact fp32) ----------
__global__ void selfpos_kernel(const float* __restrict__ hn,
                               float* __restrict__ pbuf,
                               float* __restrict__ corr) {
  int wave = threadIdx.x >> 5, lane = threadIdx.x & 31;
  int g = blockIdx.x * 8 + wave;
  int pos = (g < NNODES) ? g + NNODES : g - NNODES;
  float4 a = ((const float4*)(hn + (size_t)g * DIM))[lane];
  float4 b = ((const float4*)(hn + (size_t)pos * DIM))[lane];
  float dgg = a.x * a.x + a.y * a.y + a.z * a.z + a.w * a.w;
  float dgp = a.x * b.x + a.y * b.y + a.z * b.z + a.w * b.w;
  dgg = wave_reduce_sum(dgg);
  dgp = wave_reduce_sum(dgp);
  if (lane == 0) {
    pbuf[g] = 2.0f * dgp;
    corr[g] = __expf(2.0f * dgg - 2.0f) + __expf(2.0f * dgp - 2.0f);
  }
}

// ------- 3) graph-edge corrections (wave per edge, exact fp32) --------------
__global__ void edge_kernel(const float* __restrict__ hn,
                            const int* __restrict__ er,
                            const int* __restrict__ ec,
                            float* __restrict__ corr) {
  int wave = threadIdx.x >> 5, lane = threadIdx.x & 31;
  int e = blockIdx.x * 8 + wave;
  int r = er[e], c = ec[e];
  float4 a0 = ((const float4*)(hn + (size_t)r * DIM))[lane];
  float4 a1 = ((const float4*)(hn + (size_t)(r + NNODES) * DIM))[lane];
  float4 b0 = ((const float4*)(hn + (size_t)c * DIM))[lane];
  float4 b1 = ((const float4*)(hn + (size_t)(c + NNODES) * DIM))[lane];
  float d00 = a0.x*b0.x + a0.y*b0.y + a0.z*b0.z + a0.w*b0.w;
  float d01 = a0.x*b1.x + a0.y*b1.y + a0.z*b1.z + a0.w*b1.w;
  float d10 = a1.x*b0.x + a1.y*b0.y + a1.z*b0.z + a1.w*b0.w;
  float d11 = a1.x*b1.x + a1.y*b1.y + a1.z*b1.z + a1.w*b1.w;
  d00 = wave_reduce_sum(d00);
  d01 = wave_reduce_sum(d01);
  d10 = wave_reduce_sum(d10);
  d11 = wave_reduce_sum(d11);
  if (lane == 0) {
    atomicAdd(&corr[r],          __expf(2.0f*d00 - 2.0f) + __expf(2.0f*d01 - 2.0f));
    atomicAdd(&corr[r + NNODES], __expf(2.0f*d10 - 2.0f) + __expf(2.0f*d11 - 2.0f));
  }
}

// ------- 4) dense Gram pass: split-bf16 (hi+lo) WMMA, fused exp reduce ------
// sim = 2*dot(hn_g, hn_j); dot ~= hi.hi + hi.lo + lo.hi via 3x
// V_WMMA_F32_16X16X32_BF16 per K=32 chunk. Three INDEPENDENT accumulator
// chains per tile (c_hh / c_hl / c_lh, each 4 WMMAs deep) so the WMMA pipe
// issues back-to-back from alternating chains instead of stalling on the
// accumulator RAW; combined once at the exp step.
// A frag (16x32 bf16): lane(h=l>>4) elems 0..15 -> K=16h..16h+15 (contiguous).
// B frag (32x16 bf16): lane elems 0..15 -> K=16h..16h+15 (contiguous 32B).
__launch_bounds__(128, 1)
__global__ void gram_kernel(const __bf16* __restrict__ hhi,
                            const __bf16* __restrict__ hlo,
                            float* __restrict__ rowsum) {
  __shared__ __bf16 ldsH[TKEYS * LDSB];
  __shared__ __bf16 ldsL[TKEYS * LDSB];
  const int tid  = threadIdx.x;
  const int wave = tid >> 5, lane = tid & 31;
  const int half = lane >> 4, l15 = lane & 15;

  const int qrow = blockIdx.x * QROWS + wave * 16 + l15;
  const __bf16* rah = hhi + (size_t)qrow * DIM;
  const __bf16* ral = hlo + (size_t)qrow * DIM;

  v16bf ah[4], al[4];
#pragma unroll
  for (int k = 0; k < 4; ++k) {
    v8bf h0 = *(const v8bf*)(rah + 32 * k + 16 * half);
    v8bf h1 = *(const v8bf*)(rah + 32 * k + 8 + 16 * half);
    ah[k] = cat8(h0, h1);
    v8bf q0 = *(const v8bf*)(ral + 32 * k + 16 * half);
    v8bf q1 = *(const v8bf*)(ral + 32 * k + 8 + 16 * half);
    al[k] = cat8(q0, q1);
  }

  float acc[8];
#pragma unroll
  for (int i = 0; i < 8; ++i) acc[i] = 0.0f;

  for (int kb = 0; kb < TWO_N; kb += TKEYS) {
    __syncthreads();
    // stage 64 key rows of both bf16 planes (16KB each), padded stride
#pragma unroll
    for (int i = 0; i < (TKEYS * DIM / 8) / 128; ++i) {  // 8 iters x 8 elems
      int idx = tid + i * 128;
      int row = idx >> 4;          // 16 8-elem chunks per row
      int c8  = (idx & 15) << 3;
      *(v8bf*)(&ldsH[row * LDSB + c8]) = *(const v8bf*)(hhi + (size_t)(kb + row) * DIM + c8);
      *(v8bf*)(&ldsL[row * LDSB + c8]) = *(const v8bf*)(hlo + (size_t)(kb + row) * DIM + c8);
    }
    __syncthreads();

#pragma unroll
    for (int jt = 0; jt < TKEYS / 16; ++jt) {
      const __bf16* brH = &ldsH[(jt * 16 + l15) * LDSB + 16 * half];
      const __bf16* brL = &ldsL[(jt * 16 + l15) * LDSB + 16 * half];
      v8f chh = {}, chl = {}, clh = {};
#pragma unroll
      for (int k = 0; k < 4; ++k) {
        v16bf bh = *(const v16bf*)(brH + 32 * k);
        v16bf bl = *(const v16bf*)(brL + 32 * k);
        chh = __builtin_amdgcn_wmma_f32_16x16x32_bf16(false, ah[k], false, bh,
                                                      (short)0, chh, false, false);
        chl = __builtin_amdgcn_wmma_f32_16x16x32_bf16(false, ah[k], false, bl,
                                                      (short)0, chl, false, false);
        clh = __builtin_amdgcn_wmma_f32_16x16x32_bf16(false, al[k], false, bh,
                                                      (short)0, clh, false, false);
      }
      // sim = 2*dot; accumulate exp(sim - 2) with fixed safe shift C=2
#pragma unroll
      for (int i = 0; i < 8; ++i)
        acc[i] += __expf(2.0f * (chh[i] + chl[i] + clh[i]) - 2.0f);
    }
  }

  // C/D layout: c[i] @ lane -> (M = i + 8*(lane>>4), N = lane&15)
#pragma unroll
  for (int i = 0; i < 8; ++i) {
    float s = half_reduce_sum(acc[i]);
    if (l15 == 0)
      rowsum[blockIdx.x * QROWS + wave * 16 + i + 8 * half] = s;
  }
}

// ------- 5) final loss reduction --------------------------------------------
__global__ void finalize_kernel(const float* __restrict__ rowsum,
                                const float* __restrict__ pbuf,
                                const float* __restrict__ corr,
                                float* __restrict__ out) {
  __shared__ float red[8];
  int tid = threadIdx.x;  // 256 threads
  float local = 0.0f;
  for (int g = tid; g < TWO_N; g += 256) {
    float p = pbuf[g];
    float neg = rowsum[g] - corr[g];
    neg = fmaxf(neg, 0.0f);
    local += logf(__expf(p - 2.0f) + neg + 1e-8f) + 2.0f - p;
  }
  local = wave_reduce_sum(local);
  if ((tid & 31) == 0) red[tid >> 5] = local;
  __syncthreads();
  if (tid == 0) {
    float t = 0.0f;
    for (int i = 0; i < 8; ++i) t += red[i];
    out[0] = t / (float)TWO_N;
  }
}

extern "C" void kernel_launch(void* const* d_in, const int* in_sizes, int n_in,
                              void* d_out, int out_size, void* d_ws, size_t ws_size,
                              hipStream_t stream) {
  (void)in_sizes; (void)n_in; (void)out_size; (void)ws_size;
  const float* hv = (const float*)d_in[0];
  const float* hc = (const float*)d_in[1];
  const int*   er = (const int*)d_in[2];
  const int*   ec = (const int*)d_in[3];
  float* out = (float*)d_out;

  float*  hn     = (float*)d_ws;                     // 32768x128 f32
  __bf16* hhi    = (__bf16*)(hn + (size_t)TWO_N * DIM);
  __bf16* hlo    = hhi + (size_t)TWO_N * DIM;
  float*  rowsum = (float*)(hlo + (size_t)TWO_N * DIM);
  float*  pbuf   = rowsum + TWO_N;
  float*  corr   = pbuf + TWO_N;

  norm_kernel    <<<TWO_N / 8,     256, 0, stream>>>(hv, hc, hn, hhi, hlo);
  selfpos_kernel <<<TWO_N / 8,     256, 0, stream>>>(hn, pbuf, corr);
  edge_kernel    <<<NEDGE / 8,     256, 0, stream>>>(hn, er, ec, corr);
  gram_kernel    <<<TWO_N / QROWS, 128, 0, stream>>>(hhi, hlo, rowsum);
  finalize_kernel<<<1,             256, 0, stream>>>(rowsum, pbuf, corr, out);
}
